// MPNEncoder_48996986913346
// MI455X (gfx1250) — matrix operand order repaired and checked by reading, not compiled, not run on hardware
//
#include <hip/hip_runtime.h>
#include <stdint.h>

// ---------------- problem constants (from reference) ----------------
#define HID    256
#define HEADS  4
#define NB     120000
#define NA     60000
#define MAXNB  6
#define NMOLS  2500
#define AFD    133
#define BFD    147
#define KB0    160   // bond fdim padded to multiple of 32
#define KAO    416   // (133 + 256) padded to multiple of 32

// ---------------- WMMA types ----------------
typedef __attribute__((ext_vector_type(16))) __bf16        v16bf;
typedef __attribute__((ext_vector_type(8)))  float         v8f;
typedef __attribute__((ext_vector_type(4)))  unsigned int  v4u;

union Frag { v4u q[2]; v16bf v; };

__device__ __forceinline__ uint16_t f2bf(float f) {
  uint32_t u = __float_as_uint(f);
  uint32_t r = u + 0x7FFFu + ((u >> 16) & 1u);   // round-to-nearest-even
  return (uint16_t)(r >> 16);
}
__device__ __forceinline__ float bf2f(uint16_t h) {
  return __uint_as_float(((uint32_t)h) << 16);
}

__device__ __forceinline__ v8f wmma_bf16(const Frag& A, const Frag& B, v8f c) {
  return __builtin_amdgcn_wmma_f32_16x16x32_bf16(false, A.v, false, B.v,
                                                 (short)0, c, false, false);
}

// ---------------- cast + pad fbonds to bf16 ----------------
__global__ void k_cast_fbonds(const float* __restrict__ fb, uint16_t* __restrict__ fb16) {
  long i = (long)blockIdx.x * 256 + threadIdx.x;
  if (i >= (long)NB * KB0) return;
  int row = (int)(i / KB0), col = (int)(i % KB0);
  float v = (col < BFD) ? fb[(long)row * BFD + col] : 0.0f;
  fb16[i] = f2bf(v);
}

// ---------------- pack a weight matrix into WMMA B-fragment layout ----------------
// Logical B[k][n], Ktot = KT*32, N = NT*16.
//   transpose==0 : B[k][n] = src[k*ldsrc + n]   (zero for k >= Kreal)
//   transpose==1 : B[k][n] = src[n*ldsrc + k]
// B fragment layout (16-bit, 32x16 tile, wave32):
//   half h=lane>>4, dword j -> K=8h+2(j&3)+16(j>>2) (+0,+1), N = lane&15
// Packed: dst[((kt*NT + nt)*32 + lane)*8 + j]  -> 32B contiguous per lane per tile.
__global__ void k_packB(const float* __restrict__ src, uint32_t* __restrict__ dst,
                        int KT, int NT, int Kreal, int ldsrc, int transpose) {
  int i = blockIdx.x * 256 + threadIdx.x;
  int total = KT * NT * 32 * 8;
  if (i >= total) return;
  int j    = i & 7;
  int lane = (i >> 3) & 31;
  int nt   = (i >> 8) % NT;
  int kt   = (i >> 8) / NT;
  int n  = nt * 16 + (lane & 15);
  int k0 = kt * 32 + ((lane >> 4) << 3) + 2 * (j & 3) + ((j >> 2) << 4);
  float a = 0.0f, b = 0.0f;
  if (!transpose) {
    if (k0     < Kreal) a = src[(long)k0 * ldsrc + n];
    if (k0 + 1 < Kreal) b = src[(long)(k0 + 1) * ldsrc + n];
  } else {
    if (k0     < Kreal) a = src[(long)n * ldsrc + k0];
    if (k0 + 1 < Kreal) b = src[(long)n * ldsrc + k0 + 1];
  }
  dst[i] = (uint32_t)f2bf(a) | ((uint32_t)f2bf(b) << 16);
}

// ---------------- GEMM0: binput = fbonds @ W_i ; message0 = relu(binput) ----------------
__global__ void __launch_bounds__(256)
k_gemm_binput(const uint16_t* __restrict__ fb16, const uint32_t* __restrict__ pWi,
              float* __restrict__ binput, uint16_t* __restrict__ msg) {
  int tid = threadIdx.x, lane = tid & 31, wave = tid >> 5;
  int base = blockIdx.x * 16;
  int m = lane & 15, half = lane >> 4;
  __builtin_prefetch(fb16 + (long)(base + 16) * KB0, 0, 1);  // next M-tile
  int nt0 = wave * 2, nt1 = nt0 + 1;
  v8f acc0 = {}, acc1 = {};
  for (int kt = 0; kt < 5; ++kt) {
    Frag A, B0, B1;
    const uint16_t* ar = fb16 + (long)(base + m) * KB0 + kt * 32 + half * 8;
    A.q[0] = *(const v4u*)(ar);
    A.q[1] = *(const v4u*)(ar + 16);
    const uint32_t* bp0 = pWi + ((long)kt * 16 + nt0) * 256 + lane * 8;
    B0.q[0] = *(const v4u*)(bp0);
    B0.q[1] = *(const v4u*)(bp0 + 4);
    const uint32_t* bp1 = pWi + ((long)kt * 16 + nt1) * 256 + lane * 8;
    B1.q[0] = *(const v4u*)(bp1);
    B1.q[1] = *(const v4u*)(bp1 + 4);
    acc0 = wmma_bf16(A, B0, acc0);
    acc1 = wmma_bf16(A, B1, acc1);
  }
  // C layout: VGPR r, lanes 0-15 -> M=r ; lanes 16-31 -> M=8+r ; N = lane&15
  for (int r = 0; r < 8; ++r) {
    int row = base + r + half * 8;
    float v0 = acc0[r], v1 = acc1[r];
    binput[(long)row * HID + nt0 * 16 + m] = v0;
    binput[(long)row * HID + nt1 * 16 + m] = v1;
    msg[(long)row * HID + nt0 * 16 + m] = f2bf(fmaxf(v0, 0.0f));
    msg[(long)row * HID + nt1 * 16 + m] = f2bf(fmaxf(v1, 0.0f));
  }
}

// ---------------- fused message-passing iteration ----------------
// per block: 16 bonds. Stages:
//  1) msg rows -> LDS via GLOBAL_LOAD_ASYNC_TO_LDS_B128 (ASYNCcnt)
//  2) t[h][m][d] = msg @ W_ma[h]^T  (WMMA, bf16 into LDS; A-frags hoisted)
//  3) scores (gather-dot, 4 heads), masked softmax over k=6, comp -> LDS (overwrites t)
//  4) nei_out = comp @ W_h (WMMA, 2 accumulators share each A-frag);
//     message' = relu(binput + nei_out)
__global__ void __launch_bounds__(256)
k_mp_iter(const uint16_t* __restrict__ msgIn, uint16_t* __restrict__ msgOut,
          const float* __restrict__ binput, const int* __restrict__ bgraph,
          const uint32_t* __restrict__ pWmaT, const uint32_t* __restrict__ pWh) {
  __shared__ __align__(16) uint16_t sMsg[16 * HID];     // 8 KB
  __shared__ __align__(16) uint16_t sBig[16 * 1024];    // 32 KB: t[m][h*256+d] then comp[m][h*256+d]
  __shared__ float sScore[HEADS][16][8];                // 2 KB (k<6 used)

  int tid = threadIdx.x, lane = tid & 31, wave = tid >> 5;
  int base = blockIdx.x * 16;
  int m = lane & 15, half = lane >> 4;

  // stage 1: own 16 message rows (8 KB) -> LDS via async-to-LDS DMA path.
  // Each lane moves 16 B; 256 threads x 2 instructions = 8192 B.
  {
    unsigned long long ga = (unsigned long long)(const void*)(msgIn + (long)base * HID)
                          + (unsigned long long)tid * 16ull;
    unsigned int la = (unsigned int)(uintptr_t)(&sMsg[0]) + (unsigned int)tid * 16u;
    asm volatile("global_load_async_to_lds_b128 %0, %1, off"
                 :: "v"(la), "v"(ga) : "memory");
    asm volatile("global_load_async_to_lds_b128 %0, %1, off offset:4096"
                 :: "v"(la), "v"(ga) : "memory");
    asm volatile("s_wait_asynccnt 0x0" ::: "memory");
  }
  __syncthreads();

  // stage 2: t tiles; 64 tiles (4 heads x 16 col-tiles) over 8 waves.
  // Hoist the 8 A-fragments (shared by all 8 tiles of this wave).
  {
    Frag A[8];
    for (int kt = 0; kt < 8; ++kt) {
      const uint16_t* ar = sMsg + m * HID + kt * 32 + half * 8;
      A[kt].q[0] = *(const v4u*)(ar);
      A[kt].q[1] = *(const v4u*)(ar + 16);
    }
    for (int t = 0; t < 8; ++t) {
      int T = wave + 8 * t;
      int h = T >> 4, dt = T & 15;
      v8f acc = {};
      for (int kt = 0; kt < 8; ++kt) {
        Frag B;
        const uint32_t* bp = pWmaT + ((long)h * 128 + (long)kt * 16 + dt) * 256 + lane * 8;
        B.q[0] = *(const v4u*)(bp);
        B.q[1] = *(const v4u*)(bp + 4);
        acc = wmma_bf16(A[kt], B, acc);
      }
      for (int r = 0; r < 8; ++r) {
        int mm = r + half * 8;
        int d  = dt * 16 + m;
        sBig[mm * 1024 + h * 256 + d] = f2bf(acc[r]);
      }
    }
  }
  __syncthreads();

  // stage 3a: scores[h][m][k] = dot(t[h][m][:], nei[m][k][:]) with mask (idx!=0)
  if (tid < 96) {
    int mm = tid / 6, k = tid % 6;
    int idx = bgraph[(long)(base + mm) * MAXNB + k];
    float s0 = 0.f, s1 = 0.f, s2 = 0.f, s3 = 0.f;
    const uint16_t* nrow = msgIn + (long)idx * HID;
    const uint16_t* trow = sBig + mm * 1024;
    for (int d = 0; d < HID; d += 8) {
      v4u nq = *(const v4u*)(nrow + d);
      v4u t0 = *(const v4u*)(trow + 0 * 256 + d);
      v4u t1 = *(const v4u*)(trow + 1 * 256 + d);
      v4u t2 = *(const v4u*)(trow + 2 * 256 + d);
      v4u t3 = *(const v4u*)(trow + 3 * 256 + d);
      const uint16_t* np  = (const uint16_t*)&nq;
      const uint16_t* tp0 = (const uint16_t*)&t0;
      const uint16_t* tp1 = (const uint16_t*)&t1;
      const uint16_t* tp2 = (const uint16_t*)&t2;
      const uint16_t* tp3 = (const uint16_t*)&t3;
      for (int j = 0; j < 8; ++j) {
        float nv = bf2f(np[j]);
        s0 += nv * bf2f(tp0[j]);
        s1 += nv * bf2f(tp1[j]);
        s2 += nv * bf2f(tp2[j]);
        s3 += nv * bf2f(tp3[j]);
      }
    }
    bool valid = (idx != 0);
    sScore[0][mm][k] = valid ? s0 : -1e20f;
    sScore[1][mm][k] = valid ? s1 : -1e20f;
    sScore[2][mm][k] = valid ? s2 : -1e20f;
    sScore[3][mm][k] = valid ? s3 : -1e20f;
  }
  __syncthreads();

  // stage 3b: softmax over k=6 (64 threads: one per (h,m))
  if (tid < 64) {
    int h = tid >> 4, mm = tid & 15;
    float mx = -3.4e38f;
    for (int k = 0; k < 6; ++k) mx = fmaxf(mx, sScore[h][mm][k]);
    float e[6], sum = 0.f;
    for (int k = 0; k < 6; ++k) { e[k] = __expf(sScore[h][mm][k] - mx); sum += e[k]; }
    float inv = 1.0f / sum;
    for (int k = 0; k < 6; ++k) sScore[h][mm][k] = e[k] * inv;
  }
  __syncthreads();

  // stage 3c: comp[m][h*256+d] = sum_k w * nei  (overwrites t slots; t is dead)
  {
    int pair = tid >> 2;           // (h,m): 64 pairs x 4 d-segments
    int h = pair >> 4, mm = pair & 15;
    int dseg = (tid & 3) * 64;
    float w[6]; int idxs[6];
    for (int k = 0; k < 6; ++k) {
      w[k]    = sScore[h][mm][k];
      idxs[k] = bgraph[(long)(base + mm) * MAXNB + k];
    }
    for (int d0 = 0; d0 < 64; d0 += 8) {
      float c[8] = {0.f, 0.f, 0.f, 0.f, 0.f, 0.f, 0.f, 0.f};
      for (int k = 0; k < 6; ++k) {
        v4u nq = *(const v4u*)(msgIn + (long)idxs[k] * HID + dseg + d0);
        const uint16_t* np = (const uint16_t*)&nq;
        for (int j = 0; j < 8; ++j) c[j] += w[k] * bf2f(np[j]);
      }
      for (int j = 0; j < 8; ++j)
        sBig[mm * 1024 + h * 256 + dseg + d0 + j] = f2bf(c[j]);
    }
  }
  __syncthreads();

  // stage 4: nei_out = comp @ W_h ; message' = relu(binput + nei_out)
  {
    int nt0 = wave * 2, nt1 = nt0 + 1;
    v8f acc0 = {}, acc1 = {};
    for (int kt = 0; kt < 32; ++kt) {
      Frag A, B0, B1;
      const uint16_t* ar = sBig + m * 1024 + kt * 32 + half * 8;
      A.q[0] = *(const v4u*)(ar);
      A.q[1] = *(const v4u*)(ar + 16);
      const uint32_t* bp0 = pWh + ((long)kt * 16 + nt0) * 256 + lane * 8;
      B0.q[0] = *(const v4u*)(bp0);
      B0.q[1] = *(const v4u*)(bp0 + 4);
      const uint32_t* bp1 = pWh + ((long)kt * 16 + nt1) * 256 + lane * 8;
      B1.q[0] = *(const v4u*)(bp1);
      B1.q[1] = *(const v4u*)(bp1 + 4);
      acc0 = wmma_bf16(A, B0, acc0);
      acc1 = wmma_bf16(A, B1, acc1);
    }
    for (int r = 0; r < 8; ++r) {
      int row = base + r + half * 8;
      float v0 = binput[(long)row * HID + nt0 * 16 + m] + acc0[r];
      float v1 = binput[(long)row * HID + nt1 * 16 + m] + acc1[r];
      msgOut[(long)row * HID + nt0 * 16 + m] = f2bf(fmaxf(v0, 0.0f));
      msgOut[(long)row * HID + nt1 * 16 + m] = f2bf(fmaxf(v1, 0.0f));
    }
  }
}

// ---------------- build ainput = [fatoms | sum_k msg[agraph]] (bf16, K padded to 416) ----------------
__global__ void k_build_ainput(const float* __restrict__ fatoms, const uint16_t* __restrict__ msg,
                               const int* __restrict__ agraph, uint16_t* __restrict__ ain) {
  long i = (long)blockIdx.x * 256 + threadIdx.x;
  if (i >= (long)NA * KAO) return;
  int a = (int)(i / KAO), c = (int)(i % KAO);
  float v = 0.0f;
  if (c < AFD) {
    v = fatoms[(long)a * AFD + c];
  } else if (c < AFD + HID) {
    int d = c - AFD;
    float s = 0.0f;
    for (int k = 0; k < MAXNB; ++k) {
      int idx = agraph[(long)a * MAXNB + k];
      s += bf2f(msg[(long)idx * HID + d]);
    }
    v = s;
  }
  ain[i] = f2bf(v);
}

// ---------------- atom_h = relu(ainput @ W_o + b_o) ----------------
__global__ void __launch_bounds__(256)
k_gemm_atom(const uint16_t* __restrict__ ain, const uint32_t* __restrict__ pWo,
            const float* __restrict__ b_o, float* __restrict__ atom_h) {
  int tid = threadIdx.x, lane = tid & 31, wave = tid >> 5;
  int base = blockIdx.x * 16;
  int m = lane & 15, half = lane >> 4;
  __builtin_prefetch(ain + (long)(base + 16) * KAO, 0, 1);
  int nt0 = wave * 2, nt1 = nt0 + 1;
  v8f acc0 = {}, acc1 = {};
  for (int kt = 0; kt < 13; ++kt) {
    Frag A, B0, B1;
    const uint16_t* ar = ain + (long)(base + m) * KAO + kt * 32 + half * 8;
    A.q[0] = *(const v4u*)(ar);
    A.q[1] = *(const v4u*)(ar + 16);
    const uint32_t* bp0 = pWo + ((long)kt * 16 + nt0) * 256 + lane * 8;
    B0.q[0] = *(const v4u*)(bp0);
    B0.q[1] = *(const v4u*)(bp0 + 4);
    const uint32_t* bp1 = pWo + ((long)kt * 16 + nt1) * 256 + lane * 8;
    B1.q[0] = *(const v4u*)(bp1);
    B1.q[1] = *(const v4u*)(bp1 + 4);
    acc0 = wmma_bf16(A, B0, acc0);
    acc1 = wmma_bf16(A, B1, acc1);
  }
  float bias0 = b_o[nt0 * 16 + m];
  float bias1 = b_o[nt1 * 16 + m];
  for (int r = 0; r < 8; ++r) {
    int row = base + r + half * 8;
    atom_h[(long)row * HID + nt0 * 16 + m] = fmaxf(acc0[r] + bias0, 0.0f);
    atom_h[(long)row * HID + nt1 * 16 + m] = fmaxf(acc1[r] + bias1, 0.0f);
  }
}

// ---------------- deterministic per-molecule mean (mol_ids sorted -> binary search) ----------------
__global__ void k_pool(const float* __restrict__ atom_h, const int* __restrict__ mol_ids,
                       float* __restrict__ out) {
  int i = blockIdx.x * 256 + threadIdx.x;   // mol*256 + d
  if (i >= NMOLS * HID) return;
  int mol = i >> 8, d = i & 255;
  int lo = 0, hi = NA;
  while (lo < hi) { int mid = (lo + hi) >> 1; if (mol_ids[mid] < mol) lo = mid + 1; else hi = mid; }
  int s = lo;
  lo = s; hi = NA;
  while (lo < hi) { int mid = (lo + hi) >> 1; if (mol_ids[mid] <= mol) lo = mid + 1; else hi = mid; }
  int e = lo;
  float sum = 0.0f;
  for (int a = s; a < e; ++a) sum += atom_h[(long)a * HID + d];
  out[i] = (e > s) ? sum / (float)(e - s) : 0.0f;
}

// ---------------- launch ----------------
extern "C" void kernel_launch(void* const* d_in, const int* in_sizes, int n_in,
                              void* d_out, int out_size, void* d_ws, size_t ws_size,
                              hipStream_t stream) {
  (void)in_sizes; (void)n_in; (void)out_size; (void)ws_size;
  const float* fatoms  = (const float*)d_in[0];
  const float* fbonds  = (const float*)d_in[1];
  const int*   agraph  = (const int*)d_in[2];
  const int*   bgraph  = (const int*)d_in[3];
  const int*   mol_ids = (const int*)d_in[4];
  // d_in[5] = n_mols scalar (compile-time constant NMOLS)
  const float* W_i  = (const float*)d_in[6];
  const float* W_ma = (const float*)d_in[7];
  const float* W_h  = (const float*)d_in[8];
  const float* W_o  = (const float*)d_in[9];
  const float* b_o  = (const float*)d_in[10];
  float* out = (float*)d_out;

  size_t off = 0;
  auto take = [&](size_t bytes) -> void* {
    off = (off + 255) & ~(size_t)255;
    void* p = (char*)d_ws + off;
    off += bytes;
    return p;
  };
  float*    binput = (float*)   take((size_t)NB * HID * 4);
  uint16_t* msgA   = (uint16_t*)take((size_t)NB * HID * 2);
  uint16_t* msgB   = (uint16_t*)take((size_t)NB * HID * 2);
  uint16_t* fb16   = (uint16_t*)take((size_t)NB * KB0 * 2);
  uint16_t* ain    = (uint16_t*)take((size_t)NA * KAO * 2);
  float*    atom_h = (float*)   take((size_t)NA * HID * 4);
  uint32_t* pWi    = (uint32_t*)take((size_t)5  * 16 * 256 * 4);
  uint32_t* pWma   = (uint32_t*)take((size_t)4 * 8 * 16 * 256 * 4);
  uint32_t* pWh    = (uint32_t*)take((size_t)32 * 16 * 256 * 4);
  uint32_t* pWo    = (uint32_t*)take((size_t)13 * 16 * 256 * 4);

  // precision prep + fragment packing
  k_cast_fbonds<<<(NB * KB0 + 255) / 256, 256, 0, stream>>>(fbonds, fb16);
  k_packB<<<(5 * 16 * 256 + 255) / 256, 256, 0, stream>>>(W_i, pWi, 5, 16, BFD, 256, 0);
  for (int h = 0; h < HEADS; ++h)
    k_packB<<<(8 * 16 * 256 + 255) / 256, 256, 0, stream>>>(
        W_ma + (long)h * 256 * 256, pWma + (long)h * 8 * 16 * 256, 8, 16, 256, 256, 1);
  k_packB<<<(32 * 16 * 256 + 255) / 256, 256, 0, stream>>>(W_h, pWh, 32, 16, 1024, 256, 0);
  k_packB<<<(13 * 16 * 256 + 255) / 256, 256, 0, stream>>>(W_o, pWo, 13, 16, 389, 256, 0);

  // binput GEMM + 3 fused message-passing iterations (ping-pong)
  k_gemm_binput<<<NB / 16, 256, 0, stream>>>(fb16, pWi, binput, msgA);
  k_mp_iter<<<NB / 16, 256, 0, stream>>>(msgA, msgB, binput, bgraph, pWma, pWh);
  k_mp_iter<<<NB / 16, 256, 0, stream>>>(msgB, msgA, binput, bgraph, pWma, pWh);
  k_mp_iter<<<NB / 16, 256, 0, stream>>>(msgA, msgB, binput, bgraph, pWma, pWh);

  // atom readout + output GEMM + deterministic mean pooling
  k_build_ainput<<<(int)(((size_t)NA * KAO + 255) / 256), 256, 0, stream>>>(fatoms, msgB, agraph, ain);
  k_gemm_atom<<<NA / 16, 256, 0, stream>>>(ain, pWo, b_o, atom_h);
  k_pool<<<(NMOLS * HID + 255) / 256, 256, 0, stream>>>(atom_h, mol_ids, out);
}
